// GCN_19997367730646
// MI455X (gfx1250) — matrix operand order repaired and checked by reading, not compiled, not run on hardware
//
#include <hip/hip_runtime.h>

// GCN forward for MI455X (gfx1250, wave32, WMMA).
// Pipeline: deg -> dinv -> [WMMA GEMM -> agg-init(self-loop+bias) -> edge
// scatter(unsafeAtomicAdd f32)] x2 -> WMMA GEMM (+bias, relu-in fused).
// GEMMs use v_wmma_f32_16x16x32_f16 (f16 in / f32 acc). Edge phase is the
// bound: ~1.6 GB touched, but h/agg/x all fit in 192MB L2, so atomics run at
// L2 throughput (GLOBAL_ATOMIC_ADD_F32, coalesced thread-per-(edge,feat)).

typedef __attribute__((ext_vector_type(16))) _Float16 v16h;
typedef __attribute__((ext_vector_type(8)))  float    v8f;

#define GCN_N 100000
#define GCN_E 1600000

// ---------------- degree / normalization ----------------

__global__ __launch_bounds__(256) void gcn_deg_init(float* __restrict__ deg, int n) {
  int t = blockIdx.x * 256 + threadIdx.x;
  if (t < n) deg[t] = 1.0f;  // self-loop contributes 1 to every node's degree
}

__global__ __launch_bounds__(256) void gcn_deg_count(const long long* __restrict__ dst,
                                                     float* __restrict__ deg, int e) {
  int t = blockIdx.x * 256 + threadIdx.x;
  if (t < e) unsafeAtomicAdd(&deg[(int)dst[t]], 1.0f);
}

__global__ __launch_bounds__(256) void gcn_dinv(const float* __restrict__ deg,
                                                float* __restrict__ dinv, int n) {
  int t = blockIdx.x * 256 + threadIdx.x;
  if (t < n) {
    float d = deg[t];
    dinv[t] = d > 0.0f ? rsqrtf(d) : 0.0f;
  }
}

// ---------------- WMMA GEMM: out[M,N] = (relu?)A[M,K] @ W[K,N] (+bias) -----
// One wave computes a 16 x N band. A frag per ISA 16-bit A 16x32 table:
//   lanes 0-15 : row = lane,    K in {k0..k0+7} U {k0+16..k0+23}, k0 = kt*32
//   lanes 16-31: row = lane-16, same +8 K offset
// B frag, same K order as A:
//   lanes 0-15 : col = lane,    K in {k0..k0+7} U {k0+16..k0+23}
//   lanes 16-31: col = lane-16, same +8 K offset
// C/D: VGPR i, lanes 0-15 -> (row i, col lane); lanes 16-31 -> (row 8+i).

template <int K, int N, bool RELU_IN, bool ADD_BIAS>
__global__ __launch_bounds__(256) void gcn_gemm_wmma(const float* __restrict__ A,
                                                     const float* __restrict__ W,
                                                     const float* __restrict__ bias,
                                                     float* __restrict__ out, int M) {
  const int lane  = threadIdx.x & 31;
  const int wave  = threadIdx.x >> 5;
  const int mtile = blockIdx.x * 8 + wave;
  if (mtile * 16 >= M) return;  // uniform per wave -> EXEC stays all-ones

  const int cl    = lane & 15;
  const int hi    = lane >> 4;          // 0 for lanes 0-15, 1 for 16-31
  const int row   = mtile * 16 + cl;    // A row owned by this lane
  const int kHalfA = hi * 8;            // A K-offset per lane half
  const int kHalfB = hi * 8;            // B K-offset per lane half

  constexpr int NT = N / 16;
  constexpr int KT = K / 32;

  v8f acc[NT] = {};

#pragma unroll
  for (int kt = 0; kt < KT; ++kt) {
    // ---- A fragment (16 f16 per lane) ----
    v16h a;
    const float* arow = A + (size_t)row * K + kt * 32 + kHalfA;
#pragma unroll
    for (int j = 0; j < 8; ++j) {
      float v0 = arow[j];
      float v1 = arow[16 + j];
      if (RELU_IN) { v0 = fmaxf(v0, 0.0f); v1 = fmaxf(v1, 0.0f); }
      a[j]     = (_Float16)v0;
      a[8 + j] = (_Float16)v1;
    }
#pragma unroll
    for (int nt = 0; nt < NT; ++nt) {
      // ---- B fragment: 16 K values of column (nt*16+cl), A's K order ----
      v16h b;
      const float* wcol = W + (size_t)(kt * 32 + kHalfB) * N + nt * 16 + cl;
#pragma unroll
      for (int j = 0; j < 16; ++j) b[j] = (_Float16)wcol[(size_t)(j < 8 ? j : j + 8) * N];

      acc[nt] = __builtin_amdgcn_wmma_f32_16x16x32_f16(
          false, a, false, b, (short)0, acc[nt], false, false);
    }
  }

  // ---- store C/D ----
  const int rbase = mtile * 16 + hi * 8;
#pragma unroll
  for (int nt = 0; nt < NT; ++nt) {
    const int col = nt * 16 + cl;
    const float bv = ADD_BIAS ? bias[col] : 0.0f;
#pragma unroll
    for (int i = 0; i < 8; ++i)
      out[(size_t)(rbase + i) * N + col] = acc[nt][i] + bv;
  }
}

// ---------------- aggregation ----------------
// agg[n][f] = bias[f] + dinv[n]^2 * h[n][f]   (self-loop message + bias)
__global__ __launch_bounds__(256) void gcn_agg_init(const float* __restrict__ h,
                                                    const float* __restrict__ dinv,
                                                    const float* __restrict__ bias,
                                                    float* __restrict__ agg, int total) {
  int t = blockIdx.x * 256 + threadIdx.x;
  if (t >= total) return;
  int node = t >> 6;
  int f    = t & 63;
  float di = dinv[node];
  agg[t] = bias[f] + di * di * h[t];
}

// agg[dst][f] += dinv[src]*dinv[dst] * h[src][f], thread-per-(edge,feature).
// Consecutive lanes share an edge and cover contiguous features -> coalesced
// reads of h and coalesced GLOBAL_ATOMIC_ADD_F32 into L2.
__global__ __launch_bounds__(256) void gcn_scatter(const float* __restrict__ h,
                                                   const long long* __restrict__ src,
                                                   const long long* __restrict__ dst,
                                                   const float* __restrict__ dinv,
                                                   float* __restrict__ agg, int e) {
  long long t = (long long)blockIdx.x * 256 + threadIdx.x;
  long long ei = t >> 6;
  int f = (int)(t & 63);
  if (ei >= e) return;
  int s = (int)src[ei];
  int d = (int)dst[ei];
  float norm = dinv[s] * dinv[d];
  float msg  = norm * h[(size_t)s * 64 + f];
  unsafeAtomicAdd(&agg[(size_t)d * 64 + f], msg);
}

// ---------------- host launch ----------------

extern "C" void kernel_launch(void* const* d_in, const int* in_sizes, int n_in,
                              void* d_out, int out_size, void* d_ws, size_t ws_size,
                              hipStream_t stream) {
  const float*     x   = (const float*)d_in[0];
  const long long* ei  = (const long long*)d_in[1];   // int64 [2, E]
  const float*     W1  = (const float*)d_in[3];
  const float*     b1  = (const float*)d_in[4];
  const float*     W2  = (const float*)d_in[5];
  const float*     b2  = (const float*)d_in[6];
  const float*     Wfc = (const float*)d_in[7];
  const float*     bfc = (const float*)d_in[8];
  float*           out = (float*)d_out;

  const int N = GCN_N;
  const int E = GCN_E;
  const long long* src = ei;       // edge_index[0]
  const long long* dst = ei + E;   // edge_index[1]

  // workspace carve-out (256B aligned): deg, dinv, two 25.6MB feature buffers
  char*  ws  = (char*)d_ws;
  size_t off = 0;
  auto carve = [&](size_t bytes) {
    void* p = ws + off;
    off = (off + bytes + 255) & ~(size_t)255;
    return p;
  };
  float* deg  = (float*)carve((size_t)N * 4);
  float* dinv = (float*)carve((size_t)N * 4);
  float* bufA = (float*)carve((size_t)N * 64 * 4);   // h (layer outputs)
  float* bufB = (float*)carve((size_t)N * 64 * 4);   // agg
  (void)ws_size; (void)in_sizes; (void)n_in; (void)out_size;

  const int mtiles    = (N + 15) / 16;           // 6250
  const int gemmGrid  = (mtiles + 7) / 8;        // 8 waves (M-tiles) per block
  const int nodeGrid  = (N + 255) / 256;
  const int edgeGrid  = (E + 255) / 256;
  const int aggGrid   = (N * 64 + 255) / 256;
  const int scatGrid  = (int)(((long long)E * 64 + 255) / 256);

  // normalization
  gcn_deg_init<<<nodeGrid, 256, 0, stream>>>(deg, N);
  gcn_deg_count<<<edgeGrid, 256, 0, stream>>>(dst, deg, E);
  gcn_dinv<<<nodeGrid, 256, 0, stream>>>(deg, dinv, N);

  // layer 1: h1 = x @ W1 ; agg1 = b1 + selfloop + scatter
  gcn_gemm_wmma<128, 64, false, false><<<gemmGrid, 256, 0, stream>>>(x, W1, nullptr, bufA, N);
  gcn_agg_init<<<aggGrid, 256, 0, stream>>>(bufA, dinv, b1, bufB, N * 64);
  gcn_scatter<<<scatGrid, 256, 0, stream>>>(bufA, src, dst, dinv, bufB, E);

  // layer 2: h2 = relu(agg1) @ W2 ; agg2 = b2 + selfloop + scatter
  gcn_gemm_wmma<64, 64, true, false><<<gemmGrid, 256, 0, stream>>>(bufB, W2, nullptr, bufA, N);
  gcn_agg_init<<<aggGrid, 256, 0, stream>>>(bufA, dinv, b2, bufB, N * 64);
  gcn_scatter<<<scatGrid, 256, 0, stream>>>(bufA, src, dst, dinv, bufB, E);

  // FC: out = relu(agg2) @ Wfc + bfc
  gcn_gemm_wmma<64, 32, true, true><<<gemmGrid, 256, 0, stream>>>(bufB, Wfc, bfc, out, N);
}
